// PPN_28080496181634
// MI455X (gfx1250) — matrix operand order repaired and checked by reading, not compiled, not run on hardware
//
#include <hip/hip_runtime.h>
#include <hip/hip_bf16.h>
#include <math.h>

// ---------------------------------------------------------------------------
// Types for CDNA5 WMMA (wave32): D(16x16 f32) = A(16x32 bf16) * B(32x16 bf16) + C
// ---------------------------------------------------------------------------
typedef __bf16 v16bf __attribute__((ext_vector_type(16)));
typedef __bf16 v8bf  __attribute__((ext_vector_type(8)));
typedef float  v8f   __attribute__((ext_vector_type(8)));

// ---------------------------------------------------------------------------
// Weight conversion: f32 W[k][ci][co] -> bf16 wT[k][co_pad][ci]  (transposed,
// Cout zero-padded to a multiple of 16 so B-fragment loads need no guards)
// ---------------------------------------------------------------------------
__global__ void ppn_cvt_weightT(const float* __restrict__ W,
                                unsigned short* __restrict__ wTp,
                                int CIN, int COUT, int COUT_PAD) {
    __bf16* wT = (__bf16*)wTp;
    int total = 27 * COUT_PAD * CIN;
    for (int i = blockIdx.x * blockDim.x + threadIdx.x; i < total;
         i += gridDim.x * blockDim.x) {
        int ci = i % CIN;
        int co = (i / CIN) % COUT_PAD;
        int k  = i / (CIN * COUT_PAD);
        float v = (co < COUT) ? W[((size_t)k * CIN + ci) * COUT + co] : 0.f;
        wT[i] = (__bf16)v;
    }
}

// Fuse the 3 level-3 heads (pix:3, score:2, type:5) into one Cout=10 (pad 16) weight
__global__ void ppn_cvt_headT(const float* __restrict__ wp,
                              const float* __restrict__ wsc,
                              const float* __restrict__ wt,
                              unsigned short* __restrict__ wTp, int CIN) {
    __bf16* wT = (__bf16*)wTp;
    int total = 27 * 16 * CIN;
    for (int i = blockIdx.x * blockDim.x + threadIdx.x; i < total;
         i += gridDim.x * blockDim.x) {
        int ci = i % CIN;
        int co = (i / CIN) % 16;
        int k  = i / (CIN * 16);
        float v = 0.f;
        if (co < 3)       v = wp [((size_t)k * CIN + ci) * 3 + co];
        else if (co < 5)  v = wsc[((size_t)k * CIN + ci) * 2 + (co - 3)];
        else if (co < 10) v = wt [((size_t)k * CIN + ci) * 5 + (co - 5)];
        wT[i] = (__bf16)v;
    }
}

// Streaming f32 -> bf16 activation conversion (halves all gather traffic)
__global__ void ppn_cvt_act(const float* __restrict__ src,
                            unsigned short* __restrict__ dstp, long n) {
    __bf16* dst = (__bf16*)dstp;
    long stride = (long)gridDim.x * blockDim.x;
    for (long i = blockIdx.x * (long)blockDim.x + threadIdx.x; i < n; i += stride)
        dst[i] = (__bf16)src[i];
}

// ---------------------------------------------------------------------------
// Sparse submanifold conv as WMMA tiles. One wave computes a 16 x (16*NT)
// output tile: one A fragment (gathered rows) feeds NT WMMAs.
//   out[n, co] = sum_{k=0..26} sum_ci gather(f, nbr[n,k])[ci] * wT[k][co][ci]
// IN_BF16 : input activations already bf16, else f32 (convert at gather)
// HAS_ATT : zero gathered rows whose attention mask is 0 (mask is exactly 0/1)
// OUT_BF16: store compact bf16 [N, coutPad]; else f32 strided (ostride/ocol0/realCout)
// ---------------------------------------------------------------------------
template <int CIN, int NT, bool IN_BF16, bool HAS_ATT, bool OUT_BF16>
__global__ __launch_bounds__(256)
void ppn_subm_conv_wmma(const void* __restrict__ fin,
                        const unsigned short* __restrict__ wTp,
                        const int* __restrict__ nbr,
                        const float* __restrict__ att,
                        void* __restrict__ out,
                        int N, int coutPad, int realCout, int ostride, int ocol0) {
    const __bf16* wT = (const __bf16*)wTp;
    const int lane = threadIdx.x & 31;
    const int wid  = (blockIdx.x * blockDim.x + threadIdx.x) >> 5;
    const int tilesM = (N + 15) >> 4;
    const int tilesNG = (coutPad >> 4) / NT;       // groups of NT cout tiles
    if (wid >= tilesM * tilesNG) return;           // uniform per-wave exit
    const int tm = wid % tilesM;
    const int tg = wid / tilesM;

    const int half = lane >> 4;                    // K-half select (A & B layouts)
    const int mcol = lane & 15;                    // A row / B-and-C column
    const int m    = tm * 16 + mcol;
    const bool mvalid = (m < N);

    v8f acc[NT];
#pragma unroll
    for (int t = 0; t < NT; ++t) acc[t] = v8f{};

    const int* nrow = nbr + (size_t)(mvalid ? m : 0) * 27;
    // per-t weight column base: wT[k][cob_t][ci], lane column cob_t
    const __bf16* wcol[NT];
#pragma unroll
    for (int t = 0; t < NT; ++t)
        wcol[t] = wT + (size_t)((tg * NT + t) * 16 + mcol) * CIN + 8 * half;
    const size_t kStride = (size_t)coutPad * CIN;

    for (int k = 0; k < 27; ++k) {
        int s = mvalid ? nrow[k] : -1;
        if (HAS_ATT && s >= 0) {
            if (att[s] == 0.f) s = -1;             // mask is exactly 0 or 1
        }
        // Skip the whole 16-row block if no lane contributes (wave-uniform branch:
        // EXEC stays all-ones around the WMMAs, as the ISA requires).
        unsigned int anyrow = __builtin_amdgcn_ballot_w32(s >= 0);
        if (anyrow == 0u) continue;

#pragma unroll
        for (int kc = 0; kc < CIN / 32; ++kc) {
            // ---- A fragment: lane holds row m; half0 -> K {0..7,16..23}, half1 -> {8..15,24..31}
            v16bf a = {};
            if (s >= 0) {
                const int base = kc * 32 + 8 * half;
                if (IN_BF16) {
                    const __bf16* fr = (const __bf16*)fin + (size_t)s * CIN + base;
                    v8bf lo = *(const v8bf*)(fr);
                    v8bf hi = *(const v8bf*)(fr + 16);
#pragma unroll
                    for (int e = 0; e < 8; ++e) { a[e] = lo[e]; a[8 + e] = hi[e]; }
                } else {
                    const float* fr = (const float*)fin + (size_t)s * CIN + base;
                    float4 u0 = *(const float4*)(fr);
                    float4 u1 = *(const float4*)(fr + 4);
                    float4 u2 = *(const float4*)(fr + 16);
                    float4 u3 = *(const float4*)(fr + 20);
                    a[0]  = (__bf16)u0.x; a[1]  = (__bf16)u0.y; a[2]  = (__bf16)u0.z; a[3]  = (__bf16)u0.w;
                    a[4]  = (__bf16)u1.x; a[5]  = (__bf16)u1.y; a[6]  = (__bf16)u1.z; a[7]  = (__bf16)u1.w;
                    a[8]  = (__bf16)u2.x; a[9]  = (__bf16)u2.y; a[10] = (__bf16)u2.z; a[11] = (__bf16)u2.w;
                    a[12] = (__bf16)u3.x; a[13] = (__bf16)u3.y; a[14] = (__bf16)u3.z; a[15] = (__bf16)u3.w;
                }
            }
            // ---- NT x (B fragment + WMMA); B: half0 -> K {0..7,16..23}, half1 -> {8..15,24..31}
#pragma unroll
            for (int t = 0; t < NT; ++t) {
                const __bf16* wc = wcol[t] + (size_t)k * kStride + kc * 32;
                v8bf b0 = *(const v8bf*)(wc);
                v8bf b1 = *(const v8bf*)(wc + 16);
                v16bf b;
#pragma unroll
                for (int e = 0; e < 8; ++e) { b[e] = b0[e]; b[8 + e] = b1[e]; }
                acc[t] = __builtin_amdgcn_wmma_f32_16x16x32_bf16(
                    false, a, false, b, (short)0, acc[t], false, false);
            }
        }
    }

    // ---- store: C/D layout: lane<16 VGPR r -> M=r ; lane>=16 -> M=8+r ; N = lane&15
#pragma unroll
    for (int t = 0; t < NT; ++t) {
        const int cob = (tg * NT + t) * 16 + mcol;
        if (OUT_BF16) {
            __bf16* o = (__bf16*)out;
#pragma unroll
            for (int r = 0; r < 8; ++r) {
                int row = tm * 16 + r + 8 * half;
                if (row < N) o[(size_t)row * coutPad + cob] = (__bf16)acc[t][r];
            }
        } else {
            float* o = (float*)out;
            if (cob < realCout) {
#pragma unroll
                for (int r = 0; r < 8; ++r) {
                    int row = tm * 16 + r + 8 * half;
                    if (row < N) o[(size_t)row * ostride + ocol0 + cob] = acc[t][r];
                }
            }
        }
    }
}

// ---------------------------------------------------------------------------
// Small elementwise kernels
// ---------------------------------------------------------------------------
__global__ void ppn_write_coords(const int* __restrict__ coords,
                                 float* __restrict__ dst, int N) {
    int total = N * 4;
    for (int i = blockIdx.x * blockDim.x + threadIdx.x; i < total;
         i += gridDim.x * blockDim.x) {
        int row = i >> 2, c = i & 3;
        dst[(size_t)row * 6 + c] = (float)coords[i];
    }
}

// softmax(scores)[:,1] > 0.8 ; scores live at cols 4,5 of a stride-6 row
__global__ void ppn_mask_from_scores(const float* __restrict__ ppn,
                                     float* __restrict__ mask, int N) {
    for (int i = blockIdx.x * blockDim.x + threadIdx.x; i < N;
         i += gridDim.x * blockDim.x) {
        float s0 = ppn[(size_t)i * 6 + 4];
        float s1 = ppn[(size_t)i * 6 + 5];
        float p1 = 1.f / (1.f + __expf(s0 - s1));
        mask[i] = (p1 > 0.8f) ? 1.f : 0.f;
    }
}

__global__ void ppn_gather_parent(const float* __restrict__ mask,
                                  const int* __restrict__ parent,
                                  float* __restrict__ att, int N) {
    for (int i = blockIdx.x * blockDim.x + threadIdx.x; i < N;
         i += gridDim.x * blockDim.x) {
        att[i] = mask[parent[i]];
    }
}

// ---------------------------------------------------------------------------
// Launch
// ---------------------------------------------------------------------------
extern "C" void kernel_launch(void* const* d_in, const int* in_sizes, int n_in,
                              void* d_out, int out_size, void* d_ws, size_t ws_size,
                              hipStream_t stream) {
    const float* f1       = (const float*)d_in[0];
    const float* f2       = (const float*)d_in[1];
    const float* f3       = (const float*)d_in[2];
    const float* w1_conv  = (const float*)d_in[3];
    const float* w1_score = (const float*)d_in[4];
    const float* w2_conv  = (const float*)d_in[5];
    const float* w2_score = (const float*)d_in[6];
    const float* w3_conv  = (const float*)d_in[7];
    const float* w3_pix   = (const float*)d_in[8];
    const float* w3_score = (const float*)d_in[9];
    const float* w3_type  = (const float*)d_in[10];
    const int* coords1    = (const int*)d_in[11];
    const int* coords2    = (const int*)d_in[12];
    const int* nbr1       = (const int*)d_in[13];
    const int* nbr2       = (const int*)d_in[14];
    const int* nbr3       = (const int*)d_in[15];
    const int* parent12   = (const int*)d_in[16];
    const int* parent23   = (const int*)d_in[17];

    const int C1 = 160, C2 = 96, C3 = 32;
    const int N1 = in_sizes[0] / C1;
    const int N2 = in_sizes[1] / C2;
    const int N3 = in_sizes[2] / C3;

    // d_out: points[N3,10] | ppn1[N1,6] | ppn2[N2,6] | att[N2] | att2[N3]
    float* points = (float*)d_out;
    float* ppn1   = points + (size_t)N3 * 10;
    float* ppn2   = ppn1 + (size_t)N1 * 6;
    float* attb   = ppn2 + (size_t)N2 * 6;
    float* att2b  = attb + (size_t)N2;

    // workspace bump allocator
    char* p = (char*)d_ws;
    auto alloc = [&](size_t bytes) -> void* {
        void* r = (void*)p;
        p += (bytes + 255) & ~(size_t)255;
        return r;
    };
    auto pad256 = [](size_t b) { return (b + 255) & ~(size_t)255; };

    const size_t wBytes =
        pad256((size_t)27 * 160 * 160 * 2) + pad256((size_t)27 * 16 * 160 * 2) +
        pad256((size_t)27 * 96 * 96 * 2)  + pad256((size_t)27 * 16 * 96 * 2) +
        pad256((size_t)27 * 32 * 32 * 2)  + pad256((size_t)27 * 16 * 32 * 2);
    const size_t baseBytes = wBytes +
        pad256((size_t)N1 * C1 * 2) + pad256((size_t)N2 * C2 * 2) +
        pad256((size_t)N3 * C3 * 2) + pad256((size_t)N1 * 4) + pad256((size_t)N2 * 4);
    const size_t cvtBytes =
        pad256((size_t)N1 * C1 * 2) + pad256((size_t)N2 * C2 * 2) +
        pad256((size_t)N3 * C3 * 2);
    const bool preCvt = (baseBytes + cvtBytes) <= ws_size;  // fixed per run -> deterministic

    unsigned short* wT1c = (unsigned short*)alloc((size_t)27 * 160 * 160 * 2);
    unsigned short* wT1s = (unsigned short*)alloc((size_t)27 * 16 * 160 * 2);
    unsigned short* wT2c = (unsigned short*)alloc((size_t)27 * 96 * 96 * 2);
    unsigned short* wT2s = (unsigned short*)alloc((size_t)27 * 16 * 96 * 2);
    unsigned short* wT3c = (unsigned short*)alloc((size_t)27 * 32 * 32 * 2);
    unsigned short* wT3h = (unsigned short*)alloc((size_t)27 * 16 * 32 * 2);
    unsigned short* xb   = (unsigned short*)alloc((size_t)N1 * C1 * 2);
    unsigned short* yb   = (unsigned short*)alloc((size_t)N2 * C2 * 2);
    unsigned short* zb   = (unsigned short*)alloc((size_t)N3 * C3 * 2);
    float* mask1 = (float*)alloc((size_t)N1 * 4);
    float* mask2 = (float*)alloc((size_t)N2 * 4);
    unsigned short *f1b = nullptr, *f2b = nullptr, *f3b = nullptr;
    if (preCvt) {
        f1b = (unsigned short*)alloc((size_t)N1 * C1 * 2);
        f2b = (unsigned short*)alloc((size_t)N2 * C2 * 2);
        f3b = (unsigned short*)alloc((size_t)N3 * C3 * 2);
    }

    auto nb = [](long n) { return (unsigned)((n + 255) / 256); };
    auto convBlocks = [](int N, int coutPad, int NT) {
        long waves = (long)((N + 15) / 16) * (coutPad / 16 / NT);
        return (unsigned)((waves + 7) / 8);   // 8 waves / 256-thread block
    };

    // ---- weight prep (bf16, transposed, Cout padded to 16)
    ppn_cvt_weightT<<<nb(27L * 160 * 160), 256, 0, stream>>>(w1_conv, wT1c, 160, 160, 160);
    ppn_cvt_weightT<<<nb(27L * 16 * 160), 256, 0, stream>>>(w1_score, wT1s, 160, 2, 16);
    ppn_cvt_weightT<<<nb(27L * 96 * 96), 256, 0, stream>>>(w2_conv, wT2c, 96, 96, 96);
    ppn_cvt_weightT<<<nb(27L * 16 * 96), 256, 0, stream>>>(w2_score, wT2s, 96, 2, 16);
    ppn_cvt_weightT<<<nb(27L * 32 * 32), 256, 0, stream>>>(w3_conv, wT3c, 32, 32, 32);
    ppn_cvt_headT<<<nb(27L * 16 * 32), 256, 0, stream>>>(w3_pix, w3_score, w3_type, wT3h, 32);

    // ---- activation prep (bf16 gathers) if workspace allows
    if (preCvt) {
        ppn_cvt_act<<<nb((long)N1 * C1), 256, 0, stream>>>(f1, f1b, (long)N1 * C1);
        ppn_cvt_act<<<nb((long)N2 * C2), 256, 0, stream>>>(f2, f2b, (long)N2 * C2);
        ppn_cvt_act<<<nb((long)N3 * C3), 256, 0, stream>>>(f3, f3b, (long)N3 * C3);
    }

    // ---- level 1
    if (preCvt)
        ppn_subm_conv_wmma<160, 5, true, false, true>
            <<<convBlocks(N1, 160, 5), 256, 0, stream>>>(f1b, wT1c, nbr1, nullptr, xb, N1, 160, 160, 0, 0);
    else
        ppn_subm_conv_wmma<160, 5, false, false, true>
            <<<convBlocks(N1, 160, 5), 256, 0, stream>>>(f1, wT1c, nbr1, nullptr, xb, N1, 160, 160, 0, 0);
    ppn_subm_conv_wmma<160, 1, true, false, false>
        <<<convBlocks(N1, 16, 1), 256, 0, stream>>>(xb, wT1s, nbr1, nullptr, ppn1, N1, 16, 2, 6, 4);
    ppn_write_coords<<<nb((long)N1 * 4), 256, 0, stream>>>(coords1, ppn1, N1);
    ppn_mask_from_scores<<<nb(N1), 256, 0, stream>>>(ppn1, mask1, N1);
    ppn_gather_parent<<<nb(N2), 256, 0, stream>>>(mask1, parent12, attb, N2);

    // ---- level 2 (attention-gated)
    if (preCvt)
        ppn_subm_conv_wmma<96, 3, true, true, true>
            <<<convBlocks(N2, 96, 3), 256, 0, stream>>>(f2b, wT2c, nbr2, attb, yb, N2, 96, 96, 0, 0);
    else
        ppn_subm_conv_wmma<96, 3, false, true, true>
            <<<convBlocks(N2, 96, 3), 256, 0, stream>>>(f2, wT2c, nbr2, attb, yb, N2, 96, 96, 0, 0);
    ppn_subm_conv_wmma<96, 1, true, false, false>
        <<<convBlocks(N2, 16, 1), 256, 0, stream>>>(yb, wT2s, nbr2, nullptr, ppn2, N2, 16, 2, 6, 4);
    ppn_write_coords<<<nb((long)N2 * 4), 256, 0, stream>>>(coords2, ppn2, N2);
    ppn_mask_from_scores<<<nb(N2), 256, 0, stream>>>(ppn2, mask2, N2);
    ppn_gather_parent<<<nb(N3), 256, 0, stream>>>(mask2, parent23, att2b, N3);

    // ---- level 3 (attention-gated) + fused heads -> points[N3,10]
    if (preCvt)
        ppn_subm_conv_wmma<32, 2, true, true, true>
            <<<convBlocks(N3, 32, 2), 256, 0, stream>>>(f3b, wT3c, nbr3, att2b, zb, N3, 32, 32, 0, 0);
    else
        ppn_subm_conv_wmma<32, 2, false, true, true>
            <<<convBlocks(N3, 32, 2), 256, 0, stream>>>(f3, wT3c, nbr3, att2b, zb, N3, 32, 32, 0, 0);
    ppn_subm_conv_wmma<32, 1, true, false, false>
        <<<convBlocks(N3, 16, 1), 256, 0, stream>>>(zb, wT3h, nbr3, nullptr, points, N3, 16, 10, 10, 0);
}
